// customGRU_38414187495806
// MI455X (gfx1250) — matrix operand order repaired and checked by reading, not compiled, not run on hardware
//
#include <hip/hip_runtime.h>
#include <math.h>

#define DEVINL __device__ __forceinline__

typedef __attribute__((ext_vector_type(16))) _Float16 v16h;
typedef __attribute__((ext_vector_type(8)))  _Float16 v8h;
typedef __attribute__((ext_vector_type(8)))  float    v8f;

constexpr int B_   = 64;
constexpr int T_   = 64;
constexpr int DIN  = 512;
constexpr int H    = 1024;
constexpr int DOUT = 256;
constexpr int H3   = 3 * H;
constexpr int NWG_REC = 32;   // 32 WGs x 8 waves = 256 waves = one tile-triple each

// ---------------- WMMA helpers (CDNA5 wave32, 16x16x32 f16 -> f32) ----------

DEVINL v8f wmma16(v16h a, v16h b, v8f c) {
  return __builtin_amdgcn_wmma_f32_16x16x32_f16(
      /*neg_a=*/false, a, /*neg_b=*/false, b,
      /*c_mod=*/(short)0, c, /*reuse_a=*/false, /*reuse_b=*/false);
}

// A fragment: 16x32 f16, row-major source [*, ld]; rows m0..m0+15, cols k0..k0+31.
// Lanes 0-15: M=lane, K = {0..7, 16..23}; lanes 16-31: M=lane-16, K = {8..15, 24..31}.
DEVINL v16h load_a_h(const _Float16* __restrict__ src, int ld, int m0, int k0) {
  const int lane = threadIdx.x & 31;
  const int m  = m0 + (lane & 15);
  const int kh = (lane >> 4) << 3;              // 0 or 8
  const _Float16* p = src + (size_t)m * ld + k0 + kh;
  v8h lo = *(const v8h*)p;                      // K = k0+kh .. +7
  v8h hi = *(const v8h*)(p + 16);               // K = k0+16+kh .. +7
  v16h a;
#pragma unroll
  for (int i = 0; i < 8; ++i) { a[i] = lo[i]; a[i + 8] = hi[i]; }
  return a;
}

// Same A fragment but from f32 source with fused leaky-ReLU (for final FC).
DEVINL v16h load_a_f32_lrelu(const float* __restrict__ src, int ld, int m0, int k0) {
  const int lane = threadIdx.x & 31;
  const int m  = m0 + (lane & 15);
  const int kh = (lane >> 4) << 3;
  const float* p = src + (size_t)m * ld + k0 + kh;
  v16h a;
#pragma unroll
  for (int i = 0; i < 8; ++i) {
    float x0 = p[i], x1 = p[i + 16];
    a[i]     = (_Float16)(x0 > 0.f ? x0 : 0.01f * x0);
    a[i + 8] = (_Float16)(x1 > 0.f ? x1 : 0.01f * x1);
  }
  return a;
}

// B fragment: 32x16, B[k][n] = w[n][k] with w row-major [N, ld].
// Lanes 0-15: col n=n0+lane, K = {0..7, 16..23}; lanes 16-31: col n0+lane-16, K = {8..15, 24..31}.
DEVINL v16h load_b(const _Float16* __restrict__ w, int ld, int n0, int k0) {
  const int lane = threadIdx.x & 31;
  const int n = n0 + (lane & 15);
  const int k = k0 + ((lane >> 4) << 3);        // 0 or 8
  const _Float16* p = w + (size_t)n * ld + k;
  v8h lo = *(const v8h*)p;
  v8h hi = *(const v8h*)(p + 16);
  v16h b;
#pragma unroll
  for (int i = 0; i < 8; ++i) { b[i] = lo[i]; b[i + 8] = hi[i]; }
  return b;
}

// C/D 16x16 f32 tile mapping: VGPR v -> row v + 8*(lane>=16), col = lane&15.
DEVINL int crow(int v) { return v + (((threadIdx.x & 31) >> 4) << 3); }
DEVINL int ccol()      { return threadIdx.x & 15; }

DEVINL float sigmoidf_(float x) { return 1.0f / (1.0f + expf(-x)); }

// ---------------- device-wide barrier (persistent recurrent kernel) --------

DEVINL void grid_barrier(volatile unsigned* bar, int nwg) {
  __syncthreads();
  if (threadIdx.x == 0) {
    __threadfence();
    unsigned gen = bar[1];                       // read BEFORE arriving
    if (atomicAdd((unsigned*)&bar[0], 1u) == (unsigned)(nwg - 1)) {
      bar[0] = 0;
      __threadfence();
      atomicAdd((unsigned*)&bar[1], 1u);         // release
    } else {
      while (bar[1] == gen) { __builtin_amdgcn_s_sleep(8); }
    }
  }
  __syncthreads();
}

// ---------------- kernels ---------------------------------------------------

__global__ void cvt_f16_kernel(const float* __restrict__ s, _Float16* __restrict__ d, int n) {
  int i = blockIdx.x * blockDim.x + threadIdx.x;
  if (i < n) d[i] = (_Float16)s[i];
}

__global__ void zero_bar_kernel(unsigned* bar) {
  if (threadIdx.x < 2) bar[threadIdx.x] = 0u;
}

// xt0[b*T+t, :] = x[b,t,:] @ w_x0.T + b_x0   -- [4096 x 3072], K=512
// 2x2 register blocking: each wave owns a 32x32 output block (4 WMMA tiles),
// halving both A- and B-fragment traffic vs 1x1 tiles.
__global__ void __launch_bounds__(256) xproj_kernel(
    const _Float16* __restrict__ xh, const _Float16* __restrict__ wx0h,
    const float* __restrict__ bx0, float* __restrict__ xt0) {
  const int wave = (blockIdx.x * blockDim.x + threadIdx.x) >> 5; // 0..12287
  const int mg = wave / (H3 / 32);   // 0..127  (32-row group)
  const int ng = wave % (H3 / 32);   // 0..95   (32-col group)
  const int m0 = mg * 32, n0 = ng * 32;
  v8f a00 = {}, a01 = {}, a10 = {}, a11 = {};
#pragma unroll 2
  for (int k0 = 0; k0 < DIN; k0 += 32) {
    v16h fa0 = load_a_h(xh, DIN, m0,      k0);
    v16h fa1 = load_a_h(xh, DIN, m0 + 16, k0);
    v16h fb0 = load_b(wx0h, DIN, n0,      k0);
    v16h fb1 = load_b(wx0h, DIN, n0 + 16, k0);
    a00 = wmma16(fa0, fb0, a00);
    a01 = wmma16(fa0, fb1, a01);
    a10 = wmma16(fa1, fb0, a10);
    a11 = wmma16(fa1, fb1, a11);
  }
  const int c0 = n0 + ccol(), c1 = c0 + 16;
  const float bias0 = bx0[c0], bias1 = bx0[c1];
#pragma unroll
  for (int v = 0; v < 8; ++v) {
    const size_t r0 = (size_t)(m0 + crow(v)) * H3;
    const size_t r1 = (size_t)(m0 + 16 + crow(v)) * H3;
    xt0[r0 + c0] = a00[v] + bias0;
    xt0[r0 + c1] = a01[v] + bias1;
    xt0[r1 + c0] = a10[v] + bias0;
    xt0[r1 + c1] = a11[v] + bias1;
  }
}

// Persistent 2-layer GRU recurrence. 256 waves; ONE grid barrier per step
// (the mid-step barrier of step t+1 already orders layer1(t) vs layer1(t+1)
// and protects every ping-pong RAW/WAR pair).
// Wave->tile map: WG = 4 m-tiles x 2 j-tiles, so each weight (B) fragment is
// read by 4 co-resident waves and each h (A) fragment by 2 -> WGP$ captures
// the reuse of the dominant B traffic instead of spilling it all to L2.
__global__ void __launch_bounds__(256, 1) gru_rec_kernel(
    const float* __restrict__ xt0,
    const _Float16* __restrict__ wh0h, const float* __restrict__ bh0,
    const _Float16* __restrict__ wx1h, const float* __restrict__ bx1,
    const _Float16* __restrict__ wh1h, const float* __restrict__ bh1,
    float* __restrict__ h0f, _Float16* __restrict__ h0h,
    float* __restrict__ h1f, _Float16* __restrict__ h1h,
    unsigned* __restrict__ bar) {
  const int wib = threadIdx.x >> 5;                 // wave in block: 0..7
  const int mt  = wib >> 1;                         // 0..3  (batch tile)
  const int jt  = (blockIdx.x << 1) | (wib & 1);    // 0..63 (H column tile)
  const int m0 = mt * 16, j0 = jt * 16;
  const int col = j0 + ccol();   // this lane's H column

  // zero-init plane 0 of both h states (f32 + f16 shadows)
#pragma unroll
  for (int v = 0; v < 8; ++v) {
    const size_t idx = (size_t)(m0 + crow(v)) * H + col;
    h0f[idx] = 0.f; h1f[idx] = 0.f;
    h0h[idx] = (_Float16)0.f; h1h[idx] = (_Float16)0.f;
  }
  grid_barrier(bar, NWG_REC);

  for (int t = 0; t < T_; ++t) {
    const int rp = t & 1, wp = rp ^ 1;
    const float*    h0c  = h0f + (size_t)rp * B_ * H;
    float*          h0n  = h0f + (size_t)wp * B_ * H;
    const _Float16* h0ch = h0h + (size_t)rp * B_ * H;
    _Float16*       h0nh = h0h + (size_t)wp * B_ * H;
    const float*    h1c  = h1f + (size_t)rp * B_ * H;
    float*          h1n  = h1f + (size_t)wp * B_ * H;
    const _Float16* h1ch = h1h + (size_t)rp * B_ * H;
    _Float16*       h1nh = h1h + (size_t)wp * B_ * H;

    // ---- layer 0: gates_h = h0 @ w_h0.T  (A fragment shared across 3 gates)
    v8f ar = {}, az = {}, an = {};
#pragma unroll 2
    for (int k0 = 0; k0 < H; k0 += 32) {
      v16h a = load_a_h(h0ch, H, m0, k0);
      ar = wmma16(a, load_b(wh0h, H,         j0, k0), ar);
      az = wmma16(a, load_b(wh0h, H, H     + j0, k0), az);
      an = wmma16(a, load_b(wh0h, H, 2 * H + j0, k0), an);
    }
    {
      const float br = bh0[col], bz = bh0[H + col], bn = bh0[2 * H + col];
#pragma unroll
      for (int v = 0; v < 8; ++v) {
        const int b = m0 + crow(v);
        const float* xrow = xt0 + (size_t)(b * T_ + t) * H3;  // b_x0 already folded
        float r  = sigmoidf_(xrow[col]         + ar[v] + br);
        float z  = sigmoidf_(xrow[H + col]     + az[v] + bz);
        float nn = tanhf   (xrow[2 * H + col] + r * (an[v] + bn));
        float hv = z * h0c[(size_t)b * H + col] + (1.f - z) * nn;
        h0n [(size_t)b * H + col] = hv;
        h0nh[(size_t)b * H + col] = (_Float16)hv;
      }
    }
    grid_barrier(bar, NWG_REC);   // the ONLY barrier per step

    // ---- layer 1: gx = h0_new @ w_x1.T ; gh = h1 @ w_h1.T
    // r,z accumulate gx+gh in one register tile; n needs xn & hn separately.
    v8f cr = {}, cz = {}, cxn = {}, chn = {};
#pragma unroll 2
    for (int k0 = 0; k0 < H; k0 += 32) {
      v16h a0 = load_a_h(h0nh, H, m0, k0);
      cr  = wmma16(a0, load_b(wx1h, H,         j0, k0), cr);
      cz  = wmma16(a0, load_b(wx1h, H, H     + j0, k0), cz);
      cxn = wmma16(a0, load_b(wx1h, H, 2 * H + j0, k0), cxn);
      v16h a1 = load_a_h(h1ch, H, m0, k0);
      cr  = wmma16(a1, load_b(wh1h, H,         j0, k0), cr);
      cz  = wmma16(a1, load_b(wh1h, H, H     + j0, k0), cz);
      chn = wmma16(a1, load_b(wh1h, H, 2 * H + j0, k0), chn);
    }
    {
      const float br = bx1[col]         + bh1[col];
      const float bz = bx1[H + col]     + bh1[H + col];
      const float bxn = bx1[2 * H + col], bhn = bh1[2 * H + col];
#pragma unroll
      for (int v = 0; v < 8; ++v) {
        const int b = m0 + crow(v);
        float r  = sigmoidf_(cr[v] + br);
        float z  = sigmoidf_(cz[v] + bz);
        float nn = tanhf(cxn[v] + bxn + r * (chn[v] + bhn));
        float hv = z * h1c[(size_t)b * H + col] + (1.f - z) * nn;
        h1n [(size_t)b * H + col] = hv;
        h1nh[(size_t)b * H + col] = (_Float16)hv;
      }
    }
    // no end-of-step barrier: step t+1's mid-barrier provides the ordering
  }
  // T_ even => final states live in plane 0.
}

// out = leaky_relu(h1) @ fc_w.T + fc_b   -- [64 x 256], K=1024; lrelu fused in A load
__global__ void __launch_bounds__(256) fc_kernel(
    const float* __restrict__ h1, const _Float16* __restrict__ fcwh,
    const float* __restrict__ fcb, float* __restrict__ out) {
  const int wave = (blockIdx.x * blockDim.x + threadIdx.x) >> 5; // 0..63
  const int mt = wave >> 4, nt = wave & 15;
  const int m0 = mt * 16, n0 = nt * 16;
  v8f acc = {};
#pragma unroll 4
  for (int k0 = 0; k0 < H; k0 += 32)
    acc = wmma16(load_a_f32_lrelu(h1, H, m0, k0), load_b(fcwh, H, n0, k0), acc);
  const int col = n0 + ccol();
  const float bias = fcb[col];
#pragma unroll
  for (int v = 0; v < 8; ++v)
    out[(size_t)(m0 + crow(v)) * DOUT + col] = acc[v] + bias;
}

// d_out[16384 ..] = stack([h0, h1]) flat
__global__ void copy_h_kernel(const float* __restrict__ h0, const float* __restrict__ h1,
                              float* __restrict__ out) {
  int i = blockIdx.x * blockDim.x + threadIdx.x;
  const int n = B_ * H;
  if (i < n)           out[B_ * DOUT + i]     = h0[i];
  else if (i < 2 * n)  out[B_ * DOUT + i]     = h1[i - n];
}

// ---------------- workspace layout (bytes, 256-aligned) ---------------------
constexpr size_t OFF_XT0  = 0;                                     // 4096*3072 f32
constexpr size_t OFF_WX0H = OFF_XT0  + (size_t)B_ * T_ * H3 * 4;   // 3072*512 f16
constexpr size_t OFF_WH0H = OFF_WX0H + (size_t)H3 * DIN * 2;       // 3072*1024 f16
constexpr size_t OFF_WX1H = OFF_WH0H + (size_t)H3 * H * 2;
constexpr size_t OFF_WH1H = OFF_WX1H + (size_t)H3 * H * 2;
constexpr size_t OFF_FCWH = OFF_WH1H + (size_t)H3 * H * 2;         // 256*1024 f16
constexpr size_t OFF_XH   = OFF_FCWH + (size_t)DOUT * H * 2;       // 4096*512 f16
constexpr size_t OFF_H0F  = OFF_XH   + (size_t)B_ * T_ * DIN * 2;  // 2 planes f32
constexpr size_t OFF_H1F  = OFF_H0F  + (size_t)2 * B_ * H * 4;
constexpr size_t OFF_H0H  = OFF_H1F  + (size_t)2 * B_ * H * 4;     // 2 planes f16
constexpr size_t OFF_H1H  = OFF_H0H  + (size_t)2 * B_ * H * 2;
constexpr size_t OFF_BAR  = OFF_H1H  + (size_t)2 * B_ * H * 2;

extern "C" void kernel_launch(void* const* d_in, const int* in_sizes, int n_in,
                              void* d_out, int out_size, void* d_ws, size_t ws_size,
                              hipStream_t stream) {
  const float* x    = (const float*)d_in[0];
  const float* w_x0 = (const float*)d_in[1];
  const float* b_x0 = (const float*)d_in[2];
  const float* w_h0 = (const float*)d_in[3];
  const float* b_h0 = (const float*)d_in[4];
  const float* w_x1 = (const float*)d_in[5];
  const float* b_x1 = (const float*)d_in[6];
  const float* w_h1 = (const float*)d_in[7];
  const float* b_h1 = (const float*)d_in[8];
  const float* fc_w = (const float*)d_in[9];
  const float* fc_b = (const float*)d_in[10];
  float* out = (float*)d_out;

  char* ws = (char*)d_ws;
  float*     xt0  = (float*)    (ws + OFF_XT0);
  _Float16*  wx0h = (_Float16*) (ws + OFF_WX0H);
  _Float16*  wh0h = (_Float16*) (ws + OFF_WH0H);
  _Float16*  wx1h = (_Float16*) (ws + OFF_WX1H);
  _Float16*  wh1h = (_Float16*) (ws + OFF_WH1H);
  _Float16*  fcwh = (_Float16*) (ws + OFF_FCWH);
  _Float16*  xh   = (_Float16*) (ws + OFF_XH);
  float*     h0f  = (float*)    (ws + OFF_H0F);
  float*     h1f  = (float*)    (ws + OFF_H1F);
  _Float16*  h0h  = (_Float16*) (ws + OFF_H0H);
  _Float16*  h1h  = (_Float16*) (ws + OFF_H1H);
  unsigned*  bar  = (unsigned*) (ws + OFF_BAR);

  auto cvt = [&](const float* s, _Float16* d, int n) {
    cvt_f16_kernel<<<(n + 255) / 256, 256, 0, stream>>>(s, d, n);
  };
  cvt(x,    xh,   B_ * T_ * DIN);
  cvt(w_x0, wx0h, H3 * DIN);
  cvt(w_h0, wh0h, H3 * H);
  cvt(w_x1, wx1h, H3 * H);
  cvt(w_h1, wh1h, H3 * H);
  cvt(fc_w, fcwh, DOUT * H);

  // Input projection: 128 m-groups * 96 n-groups = 12288 waves = 1536 blocks
  xproj_kernel<<<1536, 256, 0, stream>>>(xh, wx0h, b_x0, xt0);

  zero_bar_kernel<<<1, 32, 0, stream>>>(bar);

  // Persistent recurrence: exactly NWG_REC co-resident workgroups
  gru_rec_kernel<<<NWG_REC, 256, 0, stream>>>(
      xt0, wh0h, b_h0, wx1h, b_x1, wh1h, b_h1, h0f, h0h, h1f, h1h, bar);

  // Final states in plane 0 (T even)
  fc_kernel<<<8, 256, 0, stream>>>(h1f, fcwh, fc_b, out);
  copy_h_kernel<<<(2 * B_ * H + 255) / 256, 256, 0, stream>>>(h0f, h1f, out);
}
